// GraphAttentionLayer_56599079027412
// MI455X (gfx1250) — matrix operand order repaired and checked by reading, not compiled, not run on hardware
//
#include <hip/hip_runtime.h>
#include <hip/hip_bf16.h>
#include <math.h>

typedef __attribute__((ext_vector_type(16))) __bf16 v16bf;
typedef __attribute__((ext_vector_type(8)))  __bf16 v8bf;
typedef __attribute__((ext_vector_type(8)))  float  v8f;

#define DEV static __device__ __forceinline__

constexpr int B_ = 2, N_ = 1024, D_ = 512, H_ = 8, HD_ = 64, FFN_ = 2048;
constexpr int M_ = B_ * N_;   // 2048 token rows

// ---------------- WMMA fragment helpers (wave32; this code's k = 16h+s order, same for A and B) ----------------
// A (16x32 bf16): lane l row m=l&15; lanes 0-15 K in {0..15}, lanes 16-31
// K in {16..31}  ->  two contiguous 8-elem (16B) chunks from row-major src.
DEV v16bf load_frag_row(const __bf16* __restrict__ p, int ld) {
  int l = threadIdx.x & 31, mr = l & 15, hf = l >> 4;
  const __bf16* q = p + (size_t)mr * ld + hf * 16;
  v8bf c0 = *(const v8bf*)(q);
  v8bf c1 = *(const v8bf*)(q + 8);
  return __builtin_shufflevector(c0, c1, 0, 1, 2, 3, 4, 5, 6, 7,
                                 8, 9, 10, 11, 12, 13, 14, 15);
}
// B (32x16 bf16): lane l col n=l&15; K = (l>=16?16:0)..+15 contiguous when the
// operand is stored K-contiguous per column (transposed weights / K / Vt).
DEV v16bf load_frag_col(const __bf16* __restrict__ p, int ld) {
  int l = threadIdx.x & 31, nc = l & 15, hf = l >> 4;
  const __bf16* q = p + (size_t)nc * ld + hf * 16;
  v8bf c0 = *(const v8bf*)(q);
  v8bf c1 = *(const v8bf*)(q + 8);
  return __builtin_shufflevector(c0, c1, 0, 1, 2, 3, 4, 5, 6, 7,
                                 8, 9, 10, 11, 12, 13, 14, 15);
}
// generic (gathered) loaders -- used only by the tiny RPE MLP
template <typename F>
DEV v16bf load_a_gen(F f) {
  int l = threadIdx.x & 31, mr = l & 15, hf = l >> 4;
  v16bf a;
#pragma unroll
  for (int v = 0; v < 8; ++v) {
    int kb = hf * 16 + 2 * v;
    a[2 * v] = (__bf16)f(mr, kb);
    a[2 * v + 1] = (__bf16)f(mr, kb + 1);
  }
  return a;
}
template <typename F>
DEV v16bf load_b_gen(F f) {
  int l = threadIdx.x & 31, nc = l & 15, hf = l >> 4;
  v16bf b;
#pragma unroll
  for (int v = 0; v < 8; ++v) {
    int k = hf * 16 + 2 * v;
    b[2 * v] = (__bf16)f(k, nc);
    b[2 * v + 1] = (__bf16)f(k + 1, nc);
  }
  return b;
}
DEV v8f wmma_bf16(v16bf a, v16bf b, v8f c) {
  return __builtin_amdgcn_wmma_f32_16x16x32_bf16(false, a, false, b, (short)0, c,
                                                 false, false);
}
DEV int crow(int g) { return g + ((threadIdx.x >> 4) & 1) * 8; }
DEV int ccol()      { return threadIdx.x & 15; }

// ------------- pipelined GEMM core: 16 x 64 tile per wave -------------
// pa: A base (row 0, k 0), row-major, leading dim lda
// pb: B base (col 0, k 0), K-contiguous per column, leading dim ldb
// Double-buffered in registers: the k+1 fragments are issued before the k-step
// WMMAs, so each WMMA retires on a partial s_wait_loadcnt while the next
// iteration's 10 b128 loads are in flight.  ~112 VGPRs live -> no spills.
template <int KTOT>
DEV void gemm_tile(const __bf16* __restrict__ pa, int lda,
                   const __bf16* __restrict__ pb, int ldb, v8f (&acc)[4]) {
  v16bf a = load_frag_row(pa, lda);
  v16bf b[4];
#pragma unroll
  for (int s = 0; s < 4; ++s) b[s] = load_frag_col(pb + (size_t)(s * 16) * ldb, ldb);
  for (int k0 = 0; k0 < KTOT - 32; k0 += 32) {
    v16bf an = load_frag_row(pa + k0 + 32, lda);
    v16bf bn[4];
#pragma unroll
    for (int s = 0; s < 4; ++s)
      bn[s] = load_frag_col(pb + (size_t)(s * 16) * ldb + k0 + 32, ldb);
#pragma unroll
    for (int s = 0; s < 4; ++s) acc[s] = wmma_bf16(a, b[s], acc[s]);
    a = an;
#pragma unroll
    for (int s = 0; s < 4; ++s) b[s] = bn[s];
  }
#pragma unroll
  for (int s = 0; s < 4; ++s) acc[s] = wmma_bf16(a, b[s], acc[s]);
}

// ---------------- prep kernels ----------------
__global__ void k_cvt_bf16(const float* __restrict__ x, __bf16* __restrict__ y, int n) {
  int i = blockIdx.x * blockDim.x + threadIdx.x;
  if (i < n) y[i] = (__bf16)x[i];
}
// Wt[n*K + k] = (bf16)W[k*N + n]
__global__ void k_tr(const float* __restrict__ W, __bf16* __restrict__ Wt, int K, int N) {
  int i = blockIdx.x * blockDim.x + threadIdx.x;
  if (i < K * N) {
    int k = i / N, n = i % N;
    Wt[(size_t)n * K + k] = (__bf16)W[i];
  }
}

// ---------------- QKV: xb[2048,512] @ W^T, 16x64 tile per wave ----------------
__global__ void k_qkv(const __bf16* __restrict__ xb, const __bf16* __restrict__ WqT,
                      const __bf16* __restrict__ WkT, const __bf16* __restrict__ WvT,
                      const float* __restrict__ bq, const float* __restrict__ bk,
                      const float* __restrict__ bv, __bf16* __restrict__ Qb,
                      __bf16* __restrict__ Kb, __bf16* __restrict__ Vt) {
  int id = blockIdx.x;
  int mat = id / 1024, rest = id % 1024;
  int m0 = (rest >> 3) * 16, n0 = (rest & 7) * 64;
  const __bf16* Wt = (mat == 0) ? WqT : (mat == 1) ? WkT : WvT;
  const float*  bm = (mat == 0) ? bq : (mat == 1) ? bk : bv;
  v8f acc[4] = {};
  gemm_tile<D_>(xb + (size_t)m0 * D_, D_, Wt + (size_t)n0 * D_, D_, acc);
#pragma unroll
  for (int s = 0; s < 4; ++s) {
    int nglob = n0 + s * 16 + ccol();
    int h = nglob >> 6, d = nglob & 63;
    float bias = bm[nglob];
#pragma unroll
    for (int g = 0; g < 8; ++g) {
      int m = m0 + crow(g), bb = m >> 10, i = m & (N_ - 1);
      float v = acc[s][g] + bias;
      if (mat == 2)  // V stored transposed [B,H,64,N] for the AV GEMM
        Vt[((size_t)(bb * H_ + h) * HD_ + d) * N_ + i] = (__bf16)v;
      else if (mat == 1)
        Kb[((size_t)(bb * H_ + h) * N_ + i) * HD_ + d] = (__bf16)v;
      else
        Qb[((size_t)(bb * H_ + h) * N_ + i) * HD_ + d] = (__bf16)v;
    }
  }
}

// ------------- RPE MLP: relu(rel@R1+rb1)@R2+rb2 -> logits[B,H,N,N] -------------
__global__ void k_rpe(const float* __restrict__ rel, const float* __restrict__ R1,
                      const float* __restrict__ rb1, const float* __restrict__ R2,
                      const float* __restrict__ rb2, float* __restrict__ logits) {
  __shared__ float hid[16][64];
  size_t p0 = (size_t)blockIdx.x * 16;
  for (int nt = 0; nt < 4; ++nt) {          // K padded 4->32
    v16bf a = load_a_gen([&](int m, int k) { return (k < 4) ? rel[(p0 + m) * 4 + k] : 0.f; });
    v16bf b = load_b_gen([&](int k, int n) { return (k < 4) ? R1[k * 64 + nt * 16 + n] : 0.f; });
    v8f acc = {};
    acc = wmma_bf16(a, b, acc);
    int col = nt * 16 + ccol();
#pragma unroll
    for (int g = 0; g < 8; ++g) hid[crow(g)][col] = fmaxf(acc[g] + rb1[col], 0.f);
  }
  __syncthreads();
  v8f acc = {};
  for (int k0 = 0; k0 < 64; k0 += 32) {     // [16,64] @ [64, 8(pad16)]
    v16bf a = load_a_gen([&](int m, int k) { return hid[m][k0 + k]; });
    v16bf b = load_b_gen([&](int k, int n) { return (n < 8) ? R2[(k0 + k) * 8 + n] : 0.f; });
    acc = wmma_bf16(a, b, acc);
  }
  int col = ccol();
  if (col < 8) {
#pragma unroll
    for (int g = 0; g < 8; ++g) {
      size_t p = p0 + crow(g);
      int bb = (int)(p >> 20), i = (int)((p >> 10) & (N_ - 1)), j = (int)(p & (N_ - 1));
      logits[((size_t)(bb * H_ + col) * N_ + i) * N_ + j] = acc[g] + rb2[col];
    }
  }
}

// ---------- scores: logits += scale * Q @ K^T  (16 rows x 64 cols/wave) ----------
__global__ void k_scores(const __bf16* __restrict__ Qb, const __bf16* __restrict__ Kb,
                         float* __restrict__ logits) {
  int id = blockIdx.x;
  int bh = id >> 10, it = (id >> 4) & 63, jt = id & 15;
  size_t base = (size_t)bh * N_;
  v8f acc[4] = {};
  gemm_tile<HD_>(Qb + (base + it * 16) * HD_, HD_,
                 Kb + (base + jt * 64) * HD_, HD_, acc);
  int col = ccol();
#pragma unroll
  for (int s = 0; s < 4; ++s)
#pragma unroll
    for (int g = 0; g < 8; ++g) {
      size_t idx = (base + it * 16 + crow(g)) * N_ + jt * 64 + s * 16 + col;
      logits[idx] = acc[s][g] * 0.125f + logits[idx];   // 1/sqrt(64)
    }
}

__global__ void k_softmax(const float* __restrict__ logits, __bf16* __restrict__ P) {
  __shared__ float red[256];
  int t = threadIdx.x;
  size_t row = blockIdx.x;
  const float* L = logits + row * N_;
  float mx = -1e30f;
  for (int j = t; j < N_; j += 256) mx = fmaxf(mx, L[j]);
  red[t] = mx; __syncthreads();
  for (int s = 128; s > 0; s >>= 1) { if (t < s) red[t] = fmaxf(red[t], red[t + s]); __syncthreads(); }
  mx = red[0]; __syncthreads();
  float sum = 0.f;
  for (int j = t; j < N_; j += 256) sum += expf(L[j] - mx);
  red[t] = sum; __syncthreads();
  for (int s = 128; s > 0; s >>= 1) { if (t < s) red[t] += red[t + s]; __syncthreads(); }
  float inv = 1.f / red[0];
  for (int j = t; j < N_; j += 256) P[row * N_ + j] = (__bf16)(expf(L[j] - mx) * inv);
}

// ------- attn@V: P[N,N] @ V -> AO[B*N,512]; Vt is [B,H,64,N] K-contiguous -------
__global__ void k_av(const __bf16* __restrict__ P, const __bf16* __restrict__ Vt,
                     __bf16* __restrict__ AO) {
  int id = blockIdx.x;
  int bh = id >> 6, it = id & 63;
  size_t base = (size_t)bh * N_;
  v8f acc[4] = {};
  gemm_tile<N_>(P + (base + it * 16) * N_, N_,
                Vt + (size_t)bh * HD_ * N_, N_, acc);
  int bb = bh >> 3, h = bh & 7;
#pragma unroll
  for (int s = 0; s < 4; ++s) {
    int col = s * 16 + ccol();
#pragma unroll
    for (int g = 0; g < 8; ++g) {
      int i = it * 16 + crow(g);
      AO[(size_t)(bb * N_ + i) * D_ + h * HD_ + col] = (__bf16)acc[s][g];
    }
  }
}

// ---------------- resid1 = x + AO @ Wo^T + bo ----------------
__global__ void k_oproj(const __bf16* __restrict__ AO, const __bf16* __restrict__ WoT,
                        const float* __restrict__ bo, const float* __restrict__ x,
                        float* __restrict__ resid) {
  int id = blockIdx.x;
  int m0 = (id >> 3) * 16, n0 = (id & 7) * 64;
  v8f acc[4] = {};
  gemm_tile<D_>(AO + (size_t)m0 * D_, D_, WoT + (size_t)n0 * D_, D_, acc);
#pragma unroll
  for (int s = 0; s < 4; ++s) {
    int col = n0 + s * 16 + ccol();
#pragma unroll
    for (int g = 0; g < 8; ++g) {
      size_t idx = (size_t)(m0 + crow(g)) * D_ + col;
      resid[idx] = x[idx] + acc[s][g] + bo[col];
    }
  }
}

// ---------------- LayerNorm over D=512 (+ optional bf16 copy) ----------------
__global__ void k_ln(const float* __restrict__ in, const float* __restrict__ gam,
                     const float* __restrict__ bet, float* __restrict__ out,
                     __bf16* __restrict__ outb) {
  __shared__ float s1[256], s2[256];
  int t = threadIdx.x;
  size_t row = blockIdx.x;
  const float* r = in + row * D_;
  float a = r[t], c = r[t + 256];
  s1[t] = a + c; s2[t] = a * a + c * c; __syncthreads();
  for (int s = 128; s > 0; s >>= 1) { if (t < s) { s1[t] += s1[t + s]; s2[t] += s2[t + s]; } __syncthreads(); }
  float mu = s1[0] * (1.f / D_);
  float var = s2[0] * (1.f / D_) - mu * mu;
  float rstd = rsqrtf(var + 1e-5f);
  float y0 = (a - mu) * rstd * gam[t] + bet[t];
  float y1 = (c - mu) * rstd * gam[t + 256] + bet[t + 256];
  out[row * D_ + t] = y0; out[row * D_ + t + 256] = y1;
  if (outb) { outb[row * D_ + t] = (__bf16)y0; outb[row * D_ + t + 256] = (__bf16)y1; }
}

// ---------------- hf = gelu(x1b @ F1^T + fb1) ----------------
__global__ void k_ffn1(const __bf16* __restrict__ x1b, const __bf16* __restrict__ F1T,
                       const float* __restrict__ fb1, __bf16* __restrict__ hf) {
  int id = blockIdx.x;
  int m0 = (id >> 5) * 16, n0 = (id & 31) * 64;
  v8f acc[4] = {};
  gemm_tile<D_>(x1b + (size_t)m0 * D_, D_, F1T + (size_t)n0 * D_, D_, acc);
#pragma unroll
  for (int s = 0; s < 4; ++s) {
    int col = n0 + s * 16 + ccol();
    float bias = fb1[col];
#pragma unroll
    for (int g = 0; g < 8; ++g) {
      float v = acc[s][g] + bias;
      float ge = 0.5f * v * (1.f + erff(v * 0.70710678118f));   // exact gelu
      hf[(size_t)(m0 + crow(g)) * FFN_ + col] = (__bf16)ge;
    }
  }
}

// ---------------- resid2 = x1 + hf @ F2^T + fb2 ----------------
__global__ void k_ffn2(const __bf16* __restrict__ hf, const __bf16* __restrict__ F2T,
                       const float* __restrict__ fb2, const float* __restrict__ x1,
                       float* __restrict__ resid) {
  int id = blockIdx.x;
  int m0 = (id >> 3) * 16, n0 = (id & 7) * 64;
  v8f acc[4] = {};
  gemm_tile<FFN_>(hf + (size_t)m0 * FFN_, FFN_, F2T + (size_t)n0 * FFN_, FFN_, acc);
#pragma unroll
  for (int s = 0; s < 4; ++s) {
    int col = n0 + s * 16 + ccol();
#pragma unroll
    for (int g = 0; g < 8; ++g) {
      size_t idx = (size_t)(m0 + crow(g)) * D_ + col;
      resid[idx] = x1[idx] + acc[s][g] + fb2[col];
    }
  }
}

// ---------------- launch ----------------
extern "C" void kernel_launch(void* const* d_in, const int* in_sizes, int n_in,
                              void* d_out, int out_size, void* d_ws, size_t ws_size,
                              hipStream_t stream) {
  (void)in_sizes; (void)n_in; (void)out_size; (void)ws_size;
  const float* x   = (const float*)d_in[0];
  const float* rel = (const float*)d_in[1];
  const float* Wq  = (const float*)d_in[2];  const float* bq  = (const float*)d_in[3];
  const float* Wk  = (const float*)d_in[4];  const float* bk  = (const float*)d_in[5];
  const float* Wv  = (const float*)d_in[6];  const float* bv  = (const float*)d_in[7];
  const float* Wo  = (const float*)d_in[8];  const float* bo  = (const float*)d_in[9];
  const float* R1  = (const float*)d_in[10]; const float* rb1 = (const float*)d_in[11];
  const float* R2  = (const float*)d_in[12]; const float* rb2 = (const float*)d_in[13];
  const float* g1  = (const float*)d_in[14]; const float* b1  = (const float*)d_in[15];
  const float* g2  = (const float*)d_in[16]; const float* b2  = (const float*)d_in[17];
  const float* F1  = (const float*)d_in[18]; const float* fb1 = (const float*)d_in[19];
  const float* F2  = (const float*)d_in[20]; const float* fb2 = (const float*)d_in[21];
  float* out = (float*)d_out;

  char* w = (char*)d_ws;
  size_t off = 0;
  auto alloc = [&](size_t bytes) {
    char* p = w + off;
    off += (bytes + 255) & ~(size_t)255;
    return p;
  };
  __bf16* xb     = (__bf16*)alloc((size_t)M_ * D_ * 2);
  __bf16* WqT    = (__bf16*)alloc((size_t)D_ * D_ * 2);
  __bf16* WkT    = (__bf16*)alloc((size_t)D_ * D_ * 2);
  __bf16* WvT    = (__bf16*)alloc((size_t)D_ * D_ * 2);
  __bf16* WoT    = (__bf16*)alloc((size_t)D_ * D_ * 2);
  __bf16* F1T    = (__bf16*)alloc((size_t)D_ * FFN_ * 2);
  __bf16* F2T    = (__bf16*)alloc((size_t)FFN_ * D_ * 2);
  __bf16* Qb     = (__bf16*)alloc((size_t)B_ * H_ * N_ * HD_ * 2);
  __bf16* Kb     = (__bf16*)alloc((size_t)B_ * H_ * N_ * HD_ * 2);
  __bf16* Vt     = (__bf16*)alloc((size_t)B_ * H_ * HD_ * N_ * 2);
  float*  logits = (float*) alloc((size_t)B_ * H_ * N_ * N_ * 4);   // 64 MB
  __bf16* P      = (__bf16*)alloc((size_t)B_ * H_ * N_ * N_ * 2);   // 32 MB
  __bf16* AO     = (__bf16*)alloc((size_t)M_ * D_ * 2);
  float*  res1   = (float*) alloc((size_t)M_ * D_ * 4);
  float*  x1     = (float*) alloc((size_t)M_ * D_ * 4);
  __bf16* x1b    = (__bf16*)alloc((size_t)M_ * D_ * 2);
  __bf16* hf     = (__bf16*)alloc((size_t)M_ * FFN_ * 2);
  float*  res2   = (float*) alloc((size_t)M_ * D_ * 4);

  // prep: bf16 activations + transposed bf16 weights (K-contiguous per out col)
  k_cvt_bf16<<<(M_ * D_ + 255) / 256, 256, 0, stream>>>(x, xb, M_ * D_);
  k_tr<<<(D_ * D_ + 255) / 256, 256, 0, stream>>>(Wq, WqT, D_, D_);
  k_tr<<<(D_ * D_ + 255) / 256, 256, 0, stream>>>(Wk, WkT, D_, D_);
  k_tr<<<(D_ * D_ + 255) / 256, 256, 0, stream>>>(Wv, WvT, D_, D_);
  k_tr<<<(D_ * D_ + 255) / 256, 256, 0, stream>>>(Wo, WoT, D_, D_);
  k_tr<<<(D_ * FFN_ + 255) / 256, 256, 0, stream>>>(F1, F1T, D_, FFN_);
  k_tr<<<(FFN_ * D_ + 255) / 256, 256, 0, stream>>>(F2, F2T, FFN_, D_);

  k_qkv   <<<3 * 128 * 8, 32, 0, stream>>>(xb, WqT, WkT, WvT, bq, bk, bv, Qb, Kb, Vt);
  k_rpe   <<<(B_ * N_ * N_) / 16, 32, 0, stream>>>(rel, R1, rb1, R2, rb2, logits);
  k_scores<<<B_ * H_ * 64 * 16, 32, 0, stream>>>(Qb, Kb, logits);
  k_softmax<<<B_ * H_ * N_, 256, 0, stream>>>(logits, P);
  k_av    <<<B_ * H_ * 64, 32, 0, stream>>>(P, Vt, AO);
  k_oproj <<<128 * 8, 32, 0, stream>>>(AO, WoT, bo, x, res1);
  k_ln    <<<M_, 256, 0, stream>>>(res1, g1, b1, x1, x1b);
  k_ffn1  <<<128 * 32, 32, 0, stream>>>(x1b, F1T, fb1, hf);
  k_ffn2  <<<128 * 8, 32, 0, stream>>>(hf, F2T, fb2, x1, res2);
  k_ln    <<<M_, 256, 0, stream>>>(res2, g2, b2, out, (__bf16*)nullptr);
}
